// GSA_1013612282164
// MI455X (gfx1250) — matrix operand order repaired and checked by reading, not compiled, not run on hardware
//
#include <hip/hip_runtime.h>
#include <hip/hip_bf16.h>

typedef __attribute__((ext_vector_type(16))) _Float16 v16h;
typedef __attribute__((ext_vector_type(8)))  _Float16 v8h;
typedef __attribute__((ext_vector_type(8)))  float    v8f;

#if defined(__has_builtin)
#if __has_builtin(__builtin_amdgcn_tensor_load_to_lds) && __has_builtin(__builtin_amdgcn_s_wait_tensorcnt)
#define USE_TDM 1
#endif
#endif
#ifndef USE_TDM
#define USE_TDM 0
#endif

namespace {
constexpr int kB = 2, kH = 48, kW = 48, kC = 256, kNH = 8, kHD = 32;
constexpr int kL   = kH * kW;     // 2304
constexpr int kTok = kB * kL;     // 4608
constexpr float kScale = 0.17677669529663687f;  // 1/sqrt(32)
constexpr int kMRow = 36;         // LDS mask row stride (32 data + 4 pad dwords)

__device__ __forceinline__ v8h ldv8(const _Float16* p) {
  return *reinterpret_cast<const v8h*>(p);
}
__device__ __forceinline__ v16h joinv(v8h lo, v8h hi) {
  v16h r;
#pragma unroll
  for (int e = 0; e < 8; ++e) { r[e] = lo[e]; r[8 + e] = hi[e]; }
  return r;
}

#if USE_TDM
// 2-D TDM descriptor: tile 32(f32) x 64 rows from mask (row stride 2304 f32),
// LDS padding of 4 dwords after every 32 dwords (row stride 144B in LDS).
// clang-23 toolchain: 6-arg form (g0 v4u, g1 v8i, g2 v4i, g3 v4i, g4 v8i, cpol).
__device__ __forceinline__ void tdm_load_mask(const float* gsrc, unsigned lds_off) {
  typedef unsigned int v4u __attribute__((ext_vector_type(4)));
  typedef int v4i __attribute__((ext_vector_type(4)));
  typedef int v8i __attribute__((ext_vector_type(8)));
  unsigned long long ga = (unsigned long long)(uintptr_t)gsrc;
  v4u g0;
  g0[0] = 1u;                                              // count=1, user D#
  g0[1] = lds_off;                                         // lds_addr
  g0[2] = (unsigned)(ga & 0xFFFFFFFFu);                    // global_addr[31:0]
  g0[3] = (unsigned)((ga >> 32) & 0x01FFFFFFu) | (2u << 30); // addr[56:32] | type=2
  const unsigned td0 = 2304u, td1 = 2304u * 8u, tl0 = 32u, tl1 = 64u;
  const unsigned str0 = 2304u;
  v8i g1;
  g1[0] = (int)((2u << 16) | (1u << 20) | (4u << 22) | (3u << 25)); // 4B elems, pad 4dw per 32dw
  g1[1] = (int)((td0 & 0xFFFFu) << 16);                    // tensor_dim0 lo
  g1[2] = (int)((td0 >> 16) | ((td1 & 0xFFFFu) << 16));    // td0 hi | td1 lo
  g1[3] = (int)((td1 >> 16) | (tl0 << 16));                // td1 hi | tile_dim0
  g1[4] = (int)tl1;                                        // tile_dim1 | tile_dim2=0
  g1[5] = (int)str0;                                       // tensor_dim0_stride lo
  g1[6] = 0;                                               // stride hi | dim1_stride lo
  g1[7] = 0;
  v4i z4 = {};
  v8i z8 = {};
  __builtin_amdgcn_tensor_load_to_lds(g0, g1, z4, z4, z8, 0);
}
#endif

// ---------------------------------------------------------------- pack x -> f16
__global__ void pack_x_kernel(const float* __restrict__ x, _Float16* __restrict__ xh) {
  int i = blockIdx.x * 256 + threadIdx.x;      // exactly kTok*kC threads
  xh[i] = (_Float16)x[i];
}

// ---- pack weights into WMMA B layout: [(ntile*8+kb)*32 + lane]*16 + e,
// value = w[k][n], n = ntile*16 + lane%16, k = kb*32 + (lane/16)*16 + e.
__global__ void pack_w_kernel(const float* __restrict__ wq, const float* __restrict__ wk,
                              const float* __restrict__ wv, const float* __restrict__ wo,
                              _Float16* __restrict__ wpack) {
  int i = blockIdx.x * 256 + threadIdx.x;      // 4*16*8*32*16 = 262144 threads
  int e    =  i        & 15;
  int lane = (i >> 4)  & 31;
  int kb   = (i >> 9)  &  7;
  int nt   = (i >> 12) & 15;
  int widx =  i >> 16;
  int n = nt * 16 + (lane & 15);
  int k = kb * 32 + (lane >> 4) * 16 + e;
  const float* w = (widx == 0) ? wq : (widx == 1) ? wk : (widx == 2) ? wv : wo;
  wpack[i] = (_Float16)w[k * kC + n];
}

// ---------------------------------------------------------------- QKV GEMM + RoPE
// grid (288, 3), block 128. Wave = 16 tokens x 64 out-channels.
__global__ __launch_bounds__(128) void qkv_gemm_kernel(
    const _Float16* __restrict__ xh, const _Float16* __restrict__ wpack,
    const float* __restrict__ bq, const float* __restrict__ bk, const float* __restrict__ bv,
    const float* __restrict__ sinb, const float* __restrict__ cosb,
    _Float16* __restrict__ qh, _Float16* __restrict__ kh,
    _Float16* __restrict__ vT, float* __restrict__ vf) {
  const int proj = blockIdx.y;                 // 0=q 1=k 2=v
  const int mtile = blockIdx.x;
  const int wave = threadIdx.x >> 5, lane = threadIdx.x & 31;
  const int lm = lane & 15, lu = lane >> 4;
  const int mbase = mtile * 16;

  const _Float16* wb = wpack + (size_t)proj * (kC * kC);
  const float* bias = (proj == 0) ? bq : (proj == 1) ? bk : bv;

  v8f acc[4];
#pragma unroll
  for (int t = 0; t < 4; ++t) acc[t] = (v8f){};

  const _Float16* arow = xh + (size_t)(mbase + lm) * kC;
  for (int kb = 0; kb < 8; ++kb) {
    const _Float16* ap = arow + kb * 32;
    v16h a = joinv(ldv8(ap + 16 * lu), ldv8(ap + 16 * lu + 8));
#pragma unroll
    for (int t = 0; t < 4; ++t) {
      const int ntile = wave * 4 + t;
      const _Float16* bp = wb + ((size_t)(ntile * 8 + kb) * 32 + lane) * 16;
      v16h b = joinv(ldv8(bp), ldv8(bp + 8));
      acc[t] = __builtin_amdgcn_wmma_f32_16x16x32_f16(false, a, false, b,
                                                      (short)0, acc[t], false, false);
    }
  }

#pragma unroll
  for (int t = 0; t < 4; ++t) {
    const int col = (wave * 4 + t) * 16 + lm;  // 0..255
    const int n = col >> 5, hd = col & 31;
    const float bcol = bias[col];
#pragma unroll
    for (int r = 0; r < 8; ++r) {
      const int token = mbase + r + 8 * lu;
      const int bb = token / kL, hw = token % kL;
      float val = acc[t][r] + bcol;
      if (proj == 2) {
        vf[(size_t)token * kC + col] = val;
        vT[((size_t)(bb * kNH + n) * kHD + hd) * kL + hw] = (_Float16)val;
      } else {
        if (proj == 1) val *= kScale;
        float partner = __shfl_xor(val, 1, 32);       // lane pair = channel pair
        float rot = (col & 1) ? partner : -partner;   // rot[2i]=-x[2i+1], rot[2i+1]=x[2i]
        const int sc = hw * kHD + hd;
        float o = val * cosb[sc] + rot * sinb[sc];
        _Float16* dst = (proj == 0) ? qh : kh;
        dst[((size_t)(bb * kNH + n) * kL + hw) * kHD + hd] = (_Float16)o;
      }
    }
  }
}

// ---------------------------------------------------------------- 5x5 depthwise LEPE
__global__ void dwconv_kernel(const float* __restrict__ vf, const float* __restrict__ wdw,
                              const float* __restrict__ bdw, float* __restrict__ lepe) {
  const int c = threadIdx.x, token = blockIdx.x;
  const int bb = token / kL, hw = token % kL, h = hw / kW, w = hw % kW;
  float acc = bdw[c];
#pragma unroll
  for (int p = 0; p < 5; ++p)
#pragma unroll
    for (int q = 0; q < 5; ++q) {
      int hh = h + p - 2, ww = w + q - 2;
      if (hh >= 0 && hh < kH && ww >= 0 && ww < kW)
        acc += vf[((size_t)bb * kL + hh * kW + ww) * kC + c] * wdw[(p * 5 + q) * kC + c];
    }
  lepe[(size_t)token * kC + c] = acc;
}

// ---------------------------------------------------------------- flash attention
// grid (36, 16), block 128. Wave = one 16-row Q tile of one (b,n) head.
// Mask (170MB, the dominant HBM stream) is DMA'd tile-by-tile into LDS by the
// Tensor Data Mover, double-buffered against WMMA compute.
__global__ __launch_bounds__(128) void attn_kernel(
    const _Float16* __restrict__ qh, const _Float16* __restrict__ kh,
    const _Float16* __restrict__ vT, const float* __restrict__ mask,
    float* __restrict__ attnf) {
  __shared__ alignas(16) _Float16 pst[4][16 * 40 + 16];   // padded P staging / wave
#if USE_TDM
  __shared__ alignas(16) float mtile[2][64 * kMRow];      // double-buffered mask tile
#endif
  const int bn = blockIdx.y, bb = bn >> 3, n = bn & 7;
  const int wave = threadIdx.x >> 5, lane = threadIdx.x & 31;
  const int lm = lane & 15, lu = lane >> 4;
  const int qblock = blockIdx.x * 64;
  const int qbase = qblock + wave * 16;

  const _Float16* qrow = qh + ((size_t)bn * kL + qbase + lm) * kHD;
  const v16h qa = joinv(ldv8(qrow + 16 * lu), ldv8(qrow + 16 * lu + 8));

  const _Float16* kbn = kh + (size_t)bn * kL * kHD;
  const _Float16* vbn = vT + (size_t)bn * kHD * kL;
  const float* mrow_base = mask + ((size_t)n * kL + qbase) * kL;

  float mx[8], ls[8];
  v8f o0 = (v8f){}, o1 = (v8f){};
#pragma unroll
  for (int r = 0; r < 8; ++r) { mx[r] = -3.0e38f; ls[r] = 0.0f; }

#if USE_TDM
  const float* gmask = mask + ((size_t)n * kL + qblock) * kL;  // block's 64 mask rows
  const unsigned lds_m[2] = { (unsigned)(uintptr_t)&mtile[0][0],
                              (unsigned)(uintptr_t)&mtile[1][0] };
  if (wave == 0) tdm_load_mask(gmask, lds_m[0]);               // prologue DMA
#endif

  for (int kc = 0; kc < kL; kc += 32) {
    // --- S = Q K^T for 32 keys (two 16x16 C tiles)
    const _Float16* kp0 = kbn + (size_t)(kc + lm) * kHD + 16 * lu;
    const _Float16* kp1 = kbn + (size_t)(kc + 16 + lm) * kHD + 16 * lu;
    v16h b0 = joinv(ldv8(kp0), ldv8(kp0 + 8));
    v16h b1 = joinv(ldv8(kp1), ldv8(kp1 + 8));
    v8f zc = (v8f){};
    v8f s0 = __builtin_amdgcn_wmma_f32_16x16x32_f16(false, qa, false, b0, (short)0, zc, false, false);
    v8f s1 = __builtin_amdgcn_wmma_f32_16x16x32_f16(false, qa, false, b1, (short)0, zc, false, false);

#if USE_TDM
    // --- TDM double buffer: kick off next tile, wait for current, sync block
    const int cur = (kc >> 5) & 1;
    if (wave == 0) {
      if (kc + 32 < kL) {
        tdm_load_mask(gmask + kc + 32, lds_m[cur ^ 1]);
        __builtin_amdgcn_s_wait_tensorcnt(1);   // in-order: current tile complete
      } else {
        __builtin_amdgcn_s_wait_tensorcnt(0);
      }
    }
    __syncthreads();
    const float* mbufw = &mtile[cur][(wave * 16) * kMRow];  // this wave's 16 rows
#pragma unroll
    for (int r = 0; r < 8; ++r) {
      const float* mp = mbufw + (r + 8 * lu) * kMRow;
      s0[r] += mp[lm];
      s1[r] += mp[16 + lm];
    }
    __syncthreads();   // all waves done reading before buffer is re-targeted
#else
    // --- fallback: stream mask straight from HBM (nontemporal, prefetch ahead)
#pragma unroll
    for (int r = 0; r < 8; ++r) {
      const float* mp = mrow_base + (size_t)(r + 8 * lu) * kL + kc;
      __builtin_prefetch(mp + 64, 0, 0);
      s0[r] += __builtin_nontemporal_load(mp + lm);
      s1[r] += __builtin_nontemporal_load(mp + 16 + lm);
    }
#endif

    // --- online softmax (row lives in 16 lanes of one half; xor 1/2/4/8 stays in-half)
    _Float16* st = pst[wave];
#pragma unroll
    for (int r = 0; r < 8; ++r) {
      float m = fmaxf(s0[r], s1[r]);
      m = fmaxf(m, __shfl_xor(m, 1, 32));
      m = fmaxf(m, __shfl_xor(m, 2, 32));
      m = fmaxf(m, __shfl_xor(m, 4, 32));
      m = fmaxf(m, __shfl_xor(m, 8, 32));
      const float nm = fmaxf(mx[r], m);
      const float scale = __expf(mx[r] - nm);
      const float p0 = __expf(s0[r] - nm), p1 = __expf(s1[r] - nm);
      float sum = p0 + p1;
      sum += __shfl_xor(sum, 1, 32);
      sum += __shfl_xor(sum, 2, 32);
      sum += __shfl_xor(sum, 4, 32);
      sum += __shfl_xor(sum, 8, 32);
      ls[r] = ls[r] * scale + sum;
      mx[r] = nm;
      o0[r] *= scale; o1[r] *= scale;
      st[(r + 8 * lu) * 40 + lm]      = (_Float16)p0;   // C layout -> row-major stage
      st[(r + 8 * lu) * 40 + 16 + lm] = (_Float16)p1;
    }
    asm volatile("s_wait_dscnt 0" ::: "memory");

    // --- reload P in A layout (16 rows x 32 keys)
    const _Float16* pr = st + lm * 40;
  v16h pa = joinv(*reinterpret_cast<const v8h*>(pr + 16 * lu),
                    *reinterpret_cast<const v8h*>(pr + 16 * lu + 8));

    // --- O += P @ V (V pre-transposed: contiguous B tiles)
    const _Float16* vp0 = vbn + (size_t)lm * kL + kc + 16 * lu;
    const _Float16* vp1 = vbn + (size_t)(16 + lm) * kL + kc + 16 * lu;
    v16h vb0 = joinv(ldv8(vp0), ldv8(vp0 + 8));
    v16h vb1 = joinv(ldv8(vp1), ldv8(vp1 + 8));
    o0 = __builtin_amdgcn_wmma_f32_16x16x32_f16(false, pa, false, vb0, (short)0, o0, false, false);
    o1 = __builtin_amdgcn_wmma_f32_16x16x32_f16(false, pa, false, vb1, (short)0, o1, false, false);
  }

#pragma unroll
  for (int r = 0; r < 8; ++r) {
    const float inv = 1.0f / ls[r];
    const size_t token = (size_t)bb * kL + qbase + r + 8 * lu;
    attnf[token * kC + n * kHD + lm]      = o0[r] * inv;
    attnf[token * kC + n * kHD + 16 + lm] = o1[r] * inv;
  }
  (void)mrow_base;
}

// ---------------------------------------------------------------- attn + lepe -> f16
__global__ void fuse_kernel(const float* __restrict__ attnf, const float* __restrict__ lepe,
                            _Float16* __restrict__ th) {
  int i = blockIdx.x * 256 + threadIdx.x;
  th[i] = (_Float16)(attnf[i] + lepe[i]);
}

// ---------------------------------------------------------------- output projection
__global__ __launch_bounds__(128) void outproj_kernel(
    const _Float16* __restrict__ th, const _Float16* __restrict__ wpo,
    const float* __restrict__ bo, float* __restrict__ out) {
  const int mtile = blockIdx.x;
  const int wave = threadIdx.x >> 5, lane = threadIdx.x & 31;
  const int lm = lane & 15, lu = lane >> 4;
  const int mbase = mtile * 16;

  v8f acc[4];
#pragma unroll
  for (int t = 0; t < 4; ++t) acc[t] = (v8f){};

  const _Float16* arow = th + (size_t)(mbase + lm) * kC;
  for (int kb = 0; kb < 8; ++kb) {
    const _Float16* ap = arow + kb * 32;
    v16h a = joinv(ldv8(ap + 16 * lu), ldv8(ap + 16 * lu + 8));
#pragma unroll
    for (int t = 0; t < 4; ++t) {
      const int ntile = wave * 4 + t;
      const _Float16* bp = wpo + ((size_t)(ntile * 8 + kb) * 32 + lane) * 16;
      v16h b = joinv(ldv8(bp), ldv8(bp + 8));
      acc[t] = __builtin_amdgcn_wmma_f32_16x16x32_f16(false, a, false, b,
                                                      (short)0, acc[t], false, false);
    }
  }
#pragma unroll
  for (int t = 0; t < 4; ++t) {
    const int col = (wave * 4 + t) * 16 + lm;
    const float bcol = bo[col];
#pragma unroll
    for (int r = 0; r < 8; ++r)
      out[(size_t)(mbase + r + 8 * lu) * kC + col] = acc[t][r] + bcol;
  }
}
} // namespace

extern "C" void kernel_launch(void* const* d_in, const int* in_sizes, int n_in,
                              void* d_out, int out_size, void* d_ws, size_t ws_size,
                              hipStream_t stream) {
  const float* x    = (const float*)d_in[0];
  const float* sinb = (const float*)d_in[1];
  const float* cosb = (const float*)d_in[2];
  const float* mask = (const float*)d_in[3];
  const float* wq = (const float*)d_in[4];  const float* bq = (const float*)d_in[5];
  const float* wk = (const float*)d_in[6];  const float* bk = (const float*)d_in[7];
  const float* wv = (const float*)d_in[8];  const float* bv = (const float*)d_in[9];
  const float* wdw = (const float*)d_in[10]; const float* bdw = (const float*)d_in[11];
  const float* wo = (const float*)d_in[12]; const float* bo = (const float*)d_in[13];
  float* out = (float*)d_out;

  const size_t SZ_H  = (size_t)kTok * kC * sizeof(_Float16);  // 2359296
  const size_t SZ_WP = (size_t)4 * kC * kC * sizeof(_Float16);
  const size_t SZ_F  = (size_t)kTok * kC * sizeof(float);
  char* p = (char*)d_ws;
  _Float16* xh    = (_Float16*)p;              p += SZ_H;
  _Float16* wpack = (_Float16*)p;              p += SZ_WP;
  _Float16* qh    = (_Float16*)p;              p += SZ_H;
  _Float16* khb   = (_Float16*)p;              p += SZ_H;
  _Float16* vT    = (_Float16*)p;              p += SZ_H;
  float*    vf    = (float*)p;                 p += SZ_F;
  float*    lepe  = (float*)p;                 p += SZ_F;
  float*    attnf = (float*)p;                 p += SZ_F;
  _Float16* th    = (_Float16*)p;              p += SZ_H;
  if ((size_t)(p - (char*)d_ws) > ws_size) return;  // ~26.5 MB required

  pack_x_kernel<<<kTok, 256, 0, stream>>>(x, xh);
  pack_w_kernel<<<1024, 256, 0, stream>>>(wq, wk, wv, wo, wpack);
  qkv_gemm_kernel<<<dim3(kTok / 16, 3), 128, 0, stream>>>(
      xh, wpack, bq, bk, bv, sinb, cosb, qh, khb, vT, vf);
  dwconv_kernel<<<kTok, 256, 0, stream>>>(vf, wdw, bdw, lepe);
  attn_kernel<<<dim3(kL / 64, kB * kNH), 128, 0, stream>>>(qh, khb, vT, mask, attnf);
  fuse_kernel<<<kTok, 256, 0, stream>>>(attnf, lepe, th);
  outproj_kernel<<<kTok / 16, 128, 0, stream>>>(th, wpack + 3 * kC * kC, bo, out);
}
